// RBFLayer_2748779069877
// MI455X (gfx1250) — matrix operand order repaired and checked by reading, not compiled, not run on hardware
//
#include <hip/hip_runtime.h>
#include <math.h>

// ---------------------------------------------------------------------------
// RBF layer on MI455X (gfx1250):
//   out[b,r] = exp(-0.5 * ( Xcat[b,:] . Wcat[r,:] + const[r] ))
// where Xcat = [x_hat^2 | x_hat]  (B x 2D),  Wcat = [1/s^2 | -2*c/s^2] (R x 2D),
// const[r] = sum_d c^2/s^2.  Single bf16 WMMA GEMM with hi/lo split-precision
// (3 v_wmma_f32_16x16x32_bf16 accumulations per 32-wide K block) for
// fp32-class accuracy on the fast bf16 matrix pipe.
// ---------------------------------------------------------------------------

typedef __attribute__((ext_vector_type(16))) __bf16 v16bf;
typedef __attribute__((ext_vector_type(8)))  __bf16 v8bf;
typedef __attribute__((ext_vector_type(8)))  float  v8f;

#define BDIM 2048
#define RDIM 256
#define DDIM 784
#define K2   1568   // 2*DDIM, multiple of 32 (49 K-blocks)

// -------------------- prep kernel 1: normalize inputs, emit bf16 hi/lo -----
__global__ __launch_bounds__(256) void rbf_prep_x(
    const float* __restrict__ x, __bf16* __restrict__ xhi, __bf16* __restrict__ xlo)
{
    const int b = blockIdx.x;
    const float* row = x + (size_t)b * DDIM;

    // sum of squares (block reduction: 8 waves of 32)
    float s = 0.f;
    for (int d = threadIdx.x; d < DDIM; d += 256) { float v = row[d]; s += v * v; }
    for (int off = 16; off > 0; off >>= 1) s += __shfl_xor(s, off, 32);
    __shared__ float wsum[8];
    const int wid = threadIdx.x >> 5, lane = threadIdx.x & 31;
    if (lane == 0) wsum[wid] = s;
    __syncthreads();
    if (wid == 0) {
        float t = (lane < 8) ? wsum[lane] : 0.f;
        for (int off = 4; off > 0; off >>= 1) t += __shfl_xor(t, off, 32);
        if (lane == 0) wsum[0] = t;
    }
    __syncthreads();
    const float inv = rsqrtf(fmaxf(wsum[0], 1e-12f));

    const size_t base = (size_t)b * K2;
    for (int d = threadIdx.x; d < DDIM; d += 256) {
        float xn = row[d] * inv;
        float x2 = xn * xn;
        __bf16 h0 = (__bf16)x2; __bf16 l0 = (__bf16)(x2 - (float)h0);
        __bf16 h1 = (__bf16)xn; __bf16 l1 = (__bf16)(xn - (float)h1);
        xhi[base + d]        = h0;  xlo[base + d]        = l0;   // k = d       : x^2
        xhi[base + DDIM + d] = h1;  xlo[base + DDIM + d] = l1;   // k = D + d   : x
    }
}

// -------------------- prep kernel 2: weights + per-center constant ---------
__global__ __launch_bounds__(256) void rbf_prep_w(
    const float* __restrict__ c, const float* __restrict__ sg,
    __bf16* __restrict__ whi, __bf16* __restrict__ wlo, float* __restrict__ cr)
{
    const int r = blockIdx.x;
    const float* crow = c  + (size_t)r * DDIM;
    const float* srow = sg + (size_t)r * DDIM;
    const size_t base = (size_t)r * K2;

    float acc = 0.f;
    for (int d = threadIdx.x; d < DDIM; d += 256) {
        float sv = srow[d];
        float w  = 1.0f / (sv * sv);
        float cv = crow[d];
        float v0 = w;               // multiplies x^2
        float v1 = -2.0f * cv * w;  // multiplies x
        acc += cv * cv * w;
        __bf16 h0 = (__bf16)v0; __bf16 l0 = (__bf16)(v0 - (float)h0);
        __bf16 h1 = (__bf16)v1; __bf16 l1 = (__bf16)(v1 - (float)h1);
        whi[base + d]        = h0;  wlo[base + d]        = l0;
        whi[base + DDIM + d] = h1;  wlo[base + DDIM + d] = l1;
    }
    for (int off = 16; off > 0; off >>= 1) acc += __shfl_xor(acc, off, 32);
    __shared__ float wsum[8];
    const int wid = threadIdx.x >> 5, lane = threadIdx.x & 31;
    if (lane == 0) wsum[wid] = acc;
    __syncthreads();
    if (threadIdx.x == 0) {
        float t = 0.f;
        for (int i = 0; i < 8; ++i) t += wsum[i];
        cr[r] = t;
    }
}

// -------------------- WMMA GEMM + exp epilogue -----------------------------
// Workgroup = 8 waves (4 along M x 2 along N), each wave owns one 16x16 tile.
// Grid = (B/64, R/32) = (32, 8) = 256 workgroups.
__global__ __launch_bounds__(256) void rbf_gemm(
    const __bf16* __restrict__ xhi, const __bf16* __restrict__ xlo,
    const __bf16* __restrict__ whi, const __bf16* __restrict__ wlo,
    const float*  __restrict__ cr,  float* __restrict__ out)
{
    const int wave = threadIdx.x >> 5;
    const int lane = threadIdx.x & 31;
    const int wm = wave & 3;          // 0..3 -> M
    const int wn = wave >> 2;         // 0..1 -> N
    const int m0 = blockIdx.x * 64 + wm * 16;
    const int n0 = blockIdx.y * 32 + wn * 16;
    const int l15 = lane & 15;
    const int lhi = lane >> 4;

    // A-frag (16x32 bf16): lane = row M=l15; K run 16*lhi + 0..15 (contiguous)
    const size_t arow = (size_t)(m0 + l15) * K2;
    // B-frag (32x16 bf16): lane = col N=l15; K run 16*lhi + 0..15 (contiguous)
    const size_t brow = (size_t)(n0 + l15) * K2;

    v8f acc = {};
    for (int kb = 0; kb < K2; kb += 32) {
        const int ka0 = kb + lhi * 16;
        const int ka1 = kb + lhi * 16 + 8;
        v8bf a0h = *(const v8bf*)(xhi + arow + ka0);
        v8bf a1h = *(const v8bf*)(xhi + arow + ka1);
        v8bf a0l = *(const v8bf*)(xlo + arow + ka0);
        v8bf a1l = *(const v8bf*)(xlo + arow + ka1);

        const int kbB = kb + lhi * 16;
        v16bf bh = *(const v16bf*)(whi + brow + kbB);
        v16bf bl = *(const v16bf*)(wlo + brow + kbB);

        v16bf ah, al;
        #pragma unroll
        for (int i = 0; i < 8; ++i) {
            ah[i] = a0h[i]; ah[i + 8] = a1h[i];
            al[i] = a0l[i]; al[i + 8] = a1l[i];
        }

        // split-precision accumulate: hi*hi + hi*lo + lo*hi (fp32 accumulator)
        acc = __builtin_amdgcn_wmma_f32_16x16x32_bf16(false, ah, false, bh,
                                                      (short)0, acc, false, false);
        acc = __builtin_amdgcn_wmma_f32_16x16x32_bf16(false, ah, false, bl,
                                                      (short)0, acc, false, false);
        acc = __builtin_amdgcn_wmma_f32_16x16x32_bf16(false, al, false, bh,
                                                      (short)0, acc, false, false);
    }

    // C/D layout: VGPR i -> M = m0 + i + 8*lhi, N = n0 + (lane&15)
    const int rcol = n0 + l15;
    const float cv = cr[rcol];
    #pragma unroll
    for (int i = 0; i < 8; ++i) {
        const int row = m0 + i + 8 * lhi;
        const float dist = acc[i] + cv;
        out[(size_t)row * RDIM + rcol] = __expf(-0.5f * dist);
    }
}

// -------------------- launch ----------------------------------------------
extern "C" void kernel_launch(void* const* d_in, const int* in_sizes, int n_in,
                              void* d_out, int out_size, void* d_ws, size_t ws_size,
                              hipStream_t stream)
{
    const float* inputs  = (const float*)d_in[0];   // (2048, 784) f32
    const float* centers = (const float*)d_in[1];   // (256, 784)  f32
    const float* sigmas  = (const float*)d_in[2];   // (256, 784)  f32
    float* out = (float*)d_out;                     // (2048, 256) f32

    char* ws = (char*)d_ws;
    const size_t sz_x = (size_t)BDIM * K2 * sizeof(__bf16);   // 6,422,528
    const size_t sz_w = (size_t)RDIM * K2 * sizeof(__bf16);   //   802,816
    __bf16* xhi = (__bf16*)(ws);
    __bf16* xlo = (__bf16*)(ws + sz_x);
    __bf16* whi = (__bf16*)(ws + 2 * sz_x);
    __bf16* wlo = (__bf16*)(ws + 2 * sz_x + sz_w);
    float*  cr  = (float*) (ws + 2 * sz_x + 2 * sz_w);

    rbf_prep_x<<<BDIM, 256, 0, stream>>>(inputs, xhi, xlo);
    rbf_prep_w<<<RDIM, 256, 0, stream>>>(centers, sigmas, whi, wlo, cr);

    dim3 grid(BDIM / 64, RDIM / 32);  // (32, 8) = 256 workgroups, 8 waves each
    rbf_gemm<<<grid, 256, 0, stream>>>(xhi, xlo, whi, wlo, cr, out);
}
